// BinarizedLayer_9844065042856
// MI455X (gfx1250) — matrix operand order repaired and checked by reading, not compiled, not run on hardware
//
#include <hip/hip_runtime.h>
#include <hip/hip_bf16.h>

typedef __attribute__((ext_vector_type(16))) __bf16 v16bf;
typedef __attribute__((ext_vector_type(8)))  __bf16 v8bf;
typedef __attribute__((ext_vector_type(8)))  float  v8f;

#define M_TOTAL 16384   // B*S = 4*4096
#define N_TOTAL 4096    // DOUT
#define K_TOTAL 1024    // DIN
#define TM 128
#define TN 128
#define TK 32
#define LDSK (TK + 8)   // 40 bf16 = 80 bytes/row, keeps 16B alignment, breaks bank conflicts

__global__ __launch_bounds__(256)
void binlinear_wmma_bf16(const float* __restrict__ A,     // input  [M,K] fp32
                         const float* __restrict__ W,     // weights [N,K] fp32
                         const float* __restrict__ c1p,
                         const float* __restrict__ c2p,
                         const float* __restrict__ bias,  // [N]
                         float* __restrict__ out)         // [M,N] fp32
{
    __shared__ __align__(16) __bf16 As[2][TM][LDSK];
    __shared__ __align__(16) __bf16 Bs[2][TN][LDSK];

    const int tid  = threadIdx.x;
    const int lane = tid & 31;
    const int wave = tid >> 5;        // 0..7
    const int half = lane >> 4;       // 0/1 (wave32 half)
    const int l16  = lane & 15;

    const int wm = wave & 3;          // 4 waves along M: 32 rows each
    const int wn = wave >> 2;         // 2 waves along N: 64 cols each

    const int mblk = blockIdx.y * TM;
    const int nblk = blockIdx.x * TN;

    // binarization constants (uniform)
    const float c1 = c1p[0], c2 = c2p[0];
    const float upper  = fmaxf(c1, c2);
    const float lower  = fminf(c1, c2);
    const float middle = upper - lower;
    const __bf16 lo_b = (__bf16)lower;
    const __bf16 up_b = (__bf16)upper;

    // staging: 2 threads per row, each covers 16 consecutive k (4x float4)
    const int lrow = tid >> 1;        // 0..127
    const int lk   = (tid & 1) * 16;  // 0 or 16

    v8f acc[2][4] = {};               // 8 accumulators of 8 f32 each

    auto load_stage = [&](int kt, int buf) {
        const int k0 = kt * TK + lk;
        // A tile: fp32 -> bf16
        const float4* ag = (const float4*)(A + (size_t)(mblk + lrow) * K_TOTAL + k0);
        #pragma unroll
        for (int i = 0; i < 4; ++i) {
            float4 v = ag[i];
            __bf16* dst = &As[buf][lrow][lk + 4 * i];
            dst[0] = (__bf16)v.x; dst[1] = (__bf16)v.y;
            dst[2] = (__bf16)v.z; dst[3] = (__bf16)v.w;
        }
        // B tile: binarize weights on the fly (exact two-valued bf16 operand)
        const float4* wg = (const float4*)(W + (size_t)(nblk + lrow) * K_TOTAL + k0);
        #pragma unroll
        for (int i = 0; i < 4; ++i) {
            float4 v = wg[i];
            __bf16* dst = &Bs[buf][lrow][lk + 4 * i];
            dst[0] = (v.x < middle) ? lo_b : up_b;
            dst[1] = (v.y < middle) ? lo_b : up_b;
            dst[2] = (v.z < middle) ? lo_b : up_b;
            dst[3] = (v.w < middle) ? lo_b : up_b;
        }
    };

    load_stage(0, 0);
    __syncthreads();

    const int NT = K_TOTAL / TK;      // 32 K-steps
    for (int kt = 0; kt < NT; ++kt) {
        const int cur = kt & 1;
        if (kt + 1 < NT) load_stage(kt + 1, cur ^ 1);   // prefetch into other buffer

        #pragma unroll
        for (int mi = 0; mi < 2; ++mi) {
            // A fragment (16x32 bf16): M = l16; half selects K slices {0-7,16-23} / {8-15,24-31}
            const __bf16* arow = &As[cur][wm * 32 + mi * 16 + l16][0];
            v8bf alo = *(const v8bf*)(arow + half * 8);
            v8bf ahi = *(const v8bf*)(arow + 16 + half * 8);
            v16bf afrag;
            #pragma unroll
            for (int j = 0; j < 8; ++j) { afrag[j] = alo[j]; afrag[8 + j] = ahi[j]; }

            #pragma unroll
            for (int ni = 0; ni < 4; ++ni) {
                // B fragment (32x16 bf16): N = l16; half selects K slices {0-7,16-23} / {8-15,24-31}
                const __bf16* brow = &Bs[cur][wn * 64 + ni * 16 + l16][0];
                v8bf blo = *(const v8bf*)(brow + half * 8);
                v8bf bhi = *(const v8bf*)(brow + 16 + half * 8);
                v16bf bfrag;
                #pragma unroll
                for (int j = 0; j < 8; ++j) { bfrag[j] = blo[j]; bfrag[8 + j] = bhi[j]; }

                acc[mi][ni] = __builtin_amdgcn_wmma_f32_16x16x32_bf16(
                    /*neg_a=*/false, afrag, /*neg_b=*/false, bfrag,
                    /*c_mod=*/(short)0, acc[mi][ni],
                    /*reuse_a=*/false, /*reuse_b=*/false);
            }
        }
        __syncthreads();
    }

    // Epilogue: C/D f32 16x16 layout -> VGPR r: M = r + 8*half, N = l16. Fused bias add.
    #pragma unroll
    for (int mi = 0; mi < 2; ++mi) {
        #pragma unroll
        for (int ni = 0; ni < 4; ++ni) {
            const int n = nblk + wn * 64 + ni * 16 + l16;
            const float b = bias[n];
            const int mbase = mblk + wm * 32 + mi * 16 + half * 8;
            #pragma unroll
            for (int r = 0; r < 8; ++r) {
                out[(size_t)(mbase + r) * N_TOTAL + n] = acc[mi][ni][r] + b;
            }
        }
    }
}

extern "C" void kernel_launch(void* const* d_in, const int* in_sizes, int n_in,
                              void* d_out, int out_size, void* d_ws, size_t ws_size,
                              hipStream_t stream) {
    const float* input   = (const float*)d_in[0];  // (4,4096,1024)
    const float* weights = (const float*)d_in[1];  // (4096,1024)
    const float* c1      = (const float*)d_in[2];  // (1,)
    const float* c2      = (const float*)d_in[3];  // (1,)
    const float* bias    = (const float*)d_in[4];  // (4096,)
    float* out = (float*)d_out;                    // (4,4096,4096)

    dim3 grid(N_TOTAL / TN, M_TOTAL / TM);         // (32, 128)
    binlinear_wmma_bf16<<<grid, dim3(256), 0, stream>>>(input, weights, c1, c2, bias, out);
}
